// Seq2seq_27496380629511
// MI455X (gfx1250) — matrix operand order repaired and checked by reading, not compiled, not run on hardware
//
#include <hip/hip_runtime.h>

// ---------------------------------------------------------------------------
// Types for CDNA5 WMMA (wave32, 16x16x32 f16 -> f32)
// ---------------------------------------------------------------------------
typedef _Float16 half_t;
typedef _Float16 v8h  __attribute__((ext_vector_type(8)));
typedef _Float16 v16h __attribute__((ext_vector_type(16)));
typedef float    v8f  __attribute__((ext_vector_type(8)));

union V16U { v16h v; struct { v8h lo; v8h hi; } s; };

// D = A(16x32) x B(32x16) + C ; f32 accumulate
__device__ __forceinline__ v8f wmma16x16x32(v16h a, v16h b, v8f c) {
    return __builtin_amdgcn_wmma_f32_16x16x32_f16(
        /*neg_a=*/false, a, /*neg_b=*/false, b,
        /*c_mod=*/(short)0, c, /*reuse_a=*/false, /*reuse_b=*/false);
}

// A-matrix tile load: row-major A[M,K], lda = K.
// Lane l holds row (row0 + l%16). Contiguous K order (matches B):
//   elements 0..7  = K (kb + off + 0..7)
//   elements 8..15 = K (kb + off + 8..15), off = (l>=16)*16
__device__ __forceinline__ v16h load_a_tile(const half_t* __restrict__ A, int lda,
                                            int row0, int kb, int lane) {
    int m   = row0 + (lane & 15);
    int off = (lane >> 4) * 16;
    const half_t* p = A + (size_t)m * lda + kb + off;
    V16U u;
    u.s.lo = *(const v8h*)(p);
    u.s.hi = *(const v8h*)(p + 8);
    return u.v;
}

// B-matrix tile load from Bt stored [N,K] row-major (i.e. B transposed),
// ldb = K. Lane l holds column (col0 + l%16); contiguous 16 halves at
// kb + (l>=16)*16  -> same contiguous K order as load_a_tile.
__device__ __forceinline__ v16h load_b_tile(const half_t* __restrict__ Bt, int ldb,
                                            int col0, int kb, int lane) {
    int n   = col0 + (lane & 15);
    int off = (lane >> 4) * 16;
    return *(const v16h*)(Bt + (size_t)n * ldb + kb + off);
}

__device__ __forceinline__ float sigm(float x) { return 1.0f / (1.0f + expf(-x)); }

// ---------------------------------------------------------------------------
// Model dimensions
// ---------------------------------------------------------------------------
#define DD   256
#define HH   512
#define BB   32
#define SS   128
#define TT   64
#define VTOC 32000
#define TDEC 63          // T-1 decode steps
#define MROW (TDEC*BB)   // 2016 head-GEMM rows

// ---------------------------------------------------------------------------
// f32 -> f16 converters
// ---------------------------------------------------------------------------
__global__ void cvt_f16_kernel(const float* __restrict__ s, half_t* __restrict__ d, int n) {
    int i = blockIdx.x * blockDim.x + threadIdx.x;
    if (i < n) d[i] = (half_t)s[i];
}

// dst[c*rows + r] = src[r*cols + c]   (for W_a: need W_a^T in Bt layout)
__global__ void cvt_f16_T_kernel(const float* __restrict__ s, half_t* __restrict__ d,
                                 int rows, int cols) {
    int i = blockIdx.x * blockDim.x + threadIdx.x;
    if (i < rows * cols) {
        int r = i / cols, c = i % cols;
        d[(size_t)c * rows + r] = (half_t)s[i];
    }
}

// seq0[(s*B+b)*D + j] = f16(src_emb[x[b,s]*D + j])   grid(S,B), block D
__global__ void embed_src_kernel(const int* __restrict__ x, const float* __restrict__ emb,
                                 half_t* __restrict__ seq) {
    int s = blockIdx.x, b = blockIdx.y;
    int tok = x[b * SS + s];
    seq[((size_t)(s * BB + b)) * DD + threadIdx.x] =
        (half_t)emb[(size_t)tok * DD + threadIdx.x];
}

// ---------------------------------------------------------------------------
// Generic WMMA GEMM: C[M,N] = A[M,K](f16,row-major) x Bt[N,K]^T + bias[N]
// block = 256 threads (8 waves); wave w -> N-tile (blockIdx.x*8 + w), M-tile blockIdx.y
// Prefetch is unconditional: one 64B over-read past the matrix stays inside
// the workspace and global_prefetch_b8 is speculative (non-faulting), while
// keeping the loop body branch-free so loads/WMMA software-pipeline.
// ---------------------------------------------------------------------------
__global__ __launch_bounds__(256)
void gemm_bt_kernel(const half_t* __restrict__ A, const half_t* __restrict__ Bt,
                    const float* __restrict__ bias, float* __restrict__ C,
                    int M, int N, int K) {
    int lane = threadIdx.x & 31;
    int wave = threadIdx.x >> 5;
    int nt = (blockIdx.x * 8 + wave) * 16;
    int mt = blockIdx.y * 16;
    if (nt >= N || mt >= M) return;   // wave-uniform
    int n  = nt + (lane & 15);
    int mb = mt + (lane >> 4) * 8;
    float bv = bias ? bias[n] : 0.0f;
    v8f acc;
#pragma unroll
    for (int r = 0; r < 8; ++r) acc[r] = bv;
    for (int kb = 0; kb < K; kb += 32) {
        __builtin_prefetch(Bt + (size_t)n * K + kb + 32, 0, 1);  // next K-block of B stream
        v16h a = load_a_tile(A, K, mt, kb, lane);
        v16h b = load_b_tile(Bt, K, nt, kb, lane);
        acc = wmma16x16x32(a, b, acc);
    }
#pragma unroll
    for (int r = 0; r < 8; ++r) C[(size_t)(mb + r) * N + n] = acc[r];
}

// ---------------------------------------------------------------------------
// Encoder recurrent kernel (one workgroup per direction, persistent over S).
// gates[32,1024] = pre_ih[t] (bias already folded) + h @ Whh^T
// LDS: h(f16 32x256) + gates(f32 32x1024) + c(f32 32x256) = 180224 B
// ---------------------------------------------------------------------------
__global__ __launch_bounds__(256)
void enc_rec_kernel(const float* __restrict__ pre_ih2,   // [2][S*B,1024]
                    const half_t* __restrict__ Whh2,     // [2][1024,256]
                    half_t* __restrict__ seq_out,        // [S*B,512]
                    float* __restrict__ h_fin,           // [32,512]
                    float* __restrict__ c_fin) {         // [32,512]
    extern __shared__ char smem[];
    half_t* hb    = (half_t*)smem;                             // 32*256 f16
    float*  gates = (float*)(smem + 32 * 256 * 2);             // 32*1024 f32
    float*  cb    = (float*)(smem + 32 * 256 * 2 + 32 * 1024 * 4); // 32*256 f32

    int dir = blockIdx.x;
    const float*  pih = pre_ih2 + (size_t)dir * SS * BB * 1024;
    const half_t* Whh = Whh2 + (size_t)dir * 1024 * 256;

    for (int i = threadIdx.x; i < 32 * 256; i += 256) { hb[i] = (half_t)0.0f; cb[i] = 0.0f; }
    __syncthreads();

    int lane = threadIdx.x & 31;
    int wave = threadIdx.x >> 5;

    for (int it = 0; it < SS; ++it) {
        int t = dir ? (SS - 1 - it) : it;
        const float* src = pih + (size_t)t * BB * 1024;
        // ---- gates GEMM: 2 x 64 = 128 tiles over 8 waves ----
        for (int tile = wave; tile < 128; tile += 8) {
            int mt = (tile & 1) * 16;
            int nt = (tile >> 1) * 16;
            int n  = nt + (lane & 15);
            int mb = mt + (lane >> 4) * 8;
            v8f acc;
#pragma unroll
            for (int r = 0; r < 8; ++r) acc[r] = src[(size_t)(mb + r) * 1024 + n];
            for (int kb = 0; kb < 256; kb += 32) {
                v16h a = load_a_tile(hb, 256, mt, kb, lane);
                v16h b = load_b_tile(Whh, 256, nt, kb, lane);
                acc = wmma16x16x32(a, b, acc);
            }
#pragma unroll
            for (int r = 0; r < 8; ++r) gates[(mb + r) * 1024 + n] = acc[r];
        }
        __syncthreads();
        // ---- elementwise LSTM cell: 32x256 cells ----
        for (int idx = threadIdx.x; idx < 32 * 256; idx += 256) {
            int m = idx >> 8, j = idx & 255;
            float gi = gates[m * 1024 + j];
            float gf = gates[m * 1024 + 256 + j];
            float gg = gates[m * 1024 + 512 + j];
            float go = gates[m * 1024 + 768 + j];
            float c2 = sigm(gf) * cb[idx] + sigm(gi) * tanhf(gg);
            float h2 = sigm(go) * tanhf(c2);
            cb[idx] = c2;
            hb[idx] = (half_t)h2;
            seq_out[(size_t)(t * BB + m) * HH + dir * 256 + j] = (half_t)h2;
            if (it == SS - 1) {
                h_fin[m * HH + dir * 256 + j] = h2;
                c_fin[m * HH + dir * 256 + j] = c2;
            }
        }
        __syncthreads();
    }
}

// ---------------------------------------------------------------------------
// Persistent decoder: attention + 3 stacked LSTM cells, 63 steps, 1 WG x 16 waves.
// LDS: h3(3x32x512 f16) + inp(32x1024 f16) + qc(32x512 f16) + sc(32x128 f32)
//    = 98304 + 65536 + 32768 + 16384 = 212992 B
// ---------------------------------------------------------------------------
__global__ __launch_bounds__(512)
void dec_run_kernel(const int* __restrict__ y, const float* __restrict__ tgt_emb,
                    const half_t* __restrict__ enc,      // [S*B,512] f16
                    const half_t* __restrict__ WaT,      // [512,512] = W_a^T
                    const half_t* __restrict__ Wih0,     // [2048,1024]
                    const half_t* __restrict__ Whh0,     // [2048,512]
                    const float*  __restrict__ b0,       // [2048]
                    const half_t* __restrict__ Wih12,    // [2][2048,512]
                    const half_t* __restrict__ Whh12,    // [2][2048,512]
                    const float*  __restrict__ b12,      // [2][2048]
                    const float*  __restrict__ hdec,     // [3][32][512] init
                    float* __restrict__ cdec,            // [3][32][512] live
                    float* __restrict__ gates,           // [32][2048] scratch
                    half_t* __restrict__ hTop) {         // [63*32,512] f16
    extern __shared__ char smem[];
    half_t* h3  = (half_t*)smem;                 // 3*32*512
    half_t* inp = h3 + 3 * 32 * 512;             // 32*1024
    half_t* qc  = inp + 32 * 1024;               // 32*512
    float*  sc  = (float*)(qc + 32 * 512);       // 32*128

    int tid = threadIdx.x, lane = tid & 31, wave = tid >> 5;

    for (int i = tid; i < 3 * 32 * 512; i += 512) h3[i] = (half_t)hdec[i];
    __syncthreads();

    for (int t = 0; t < TDEC; ++t) {
        // ---- q = h_top @ W_a  (M=32,N=512,K=512): 64 tiles / 16 waves ----
        const half_t* htop = h3 + 2 * 32 * 512;
        for (int tile = wave; tile < 64; tile += 16) {
            int mt = (tile & 1) * 16, nt = (tile >> 1) * 16;
            int n = nt + (lane & 15), mb = mt + (lane >> 4) * 8;
            v8f acc = {};
            for (int kb = 0; kb < 512; kb += 32) {
                acc = wmma16x16x32(load_a_tile(htop, 512, mt, kb, lane),
                                   load_b_tile(WaT, 512, nt, kb, lane), acc);
            }
#pragma unroll
            for (int r = 0; r < 8; ++r) qc[(mb + r) * 512 + n] = (half_t)acc[r];
        }
        __syncthreads();
        // ---- scores[b,s] = q[b] . enc[s,b] ----
        for (int p = tid; p < 32 * 128; p += 512) {
            int b = p >> 7, s = p & 127;
            const half_t* e = enc + (size_t)(s * BB + b) * HH;
            const half_t* q = qc + b * HH;
            float a = 0.0f;
            for (int k = 0; k < HH; ++k) a += (float)q[k] * (float)e[k];
            sc[b * 128 + s] = a;
        }
        __syncthreads();
        // ---- softmax per batch row ----
        if (tid < 32) {
            float mx = -1e30f;
            for (int s = 0; s < 128; ++s) mx = fmaxf(mx, sc[tid * 128 + s]);
            float sum = 0.0f;
            for (int s = 0; s < 128; ++s) { float v = expf(sc[tid * 128 + s] - mx); sc[tid * 128 + s] = v; sum += v; }
            float inv = 1.0f / sum;
            for (int s = 0; s < 128; ++s) sc[tid * 128 + s] *= inv;
        }
        __syncthreads();
        // ---- ctx + embedding into inp = [emb | ctx] ----
        for (int p = tid; p < 32 * 512; p += 512) {
            int b = p >> 9, e_ = p & 511;
            float a = 0.0f;
            for (int s = 0; s < 128; ++s)
                a += sc[b * 128 + s] * (float)enc[(size_t)(s * BB + b) * HH + e_];
            inp[b * 1024 + 512 + e_] = (half_t)a;
            int tok = y[b * TT + t];
            inp[b * 1024 + e_] = (half_t)tgt_emb[(size_t)tok * HH + e_];
        }
        __syncthreads();
        // ---- 3 stacked LSTM layers ----
        for (int l = 0; l < 3; ++l) {
            const half_t* Aih = (l == 0) ? inp : (h3 + (size_t)(l - 1) * 32 * 512);
            int Kih = (l == 0) ? 1024 : 512;
            const half_t* Wih = (l == 0) ? Wih0 : Wih12 + (size_t)(l - 1) * 2048 * 512;
            const half_t* Whh = (l == 0) ? Whh0 : Whh12 + (size_t)(l - 1) * 2048 * 512;
            const float*  bia = (l == 0) ? b0   : b12 + (l - 1) * 2048;
            const half_t* hl  = h3 + (size_t)l * 32 * 512;
            // gates = bias + Aih@Wih^T + h@Whh^T : 2x128=256 tiles / 16 waves
            for (int tile = wave; tile < 256; tile += 16) {
                int mt = (tile & 1) * 16, nt = (tile >> 1) * 16;
                int n = nt + (lane & 15), mb = mt + (lane >> 4) * 8;
                float bv = bia[n];
                v8f acc;
#pragma unroll
                for (int r = 0; r < 8; ++r) acc[r] = bv;
                for (int kb = 0; kb < Kih; kb += 32)
                    acc = wmma16x16x32(load_a_tile(Aih, Kih, mt, kb, lane),
                                       load_b_tile(Wih, Kih, nt, kb, lane), acc);
                for (int kb = 0; kb < 512; kb += 32)
                    acc = wmma16x16x32(load_a_tile(hl, 512, mt, kb, lane),
                                       load_b_tile(Whh, 512, nt, kb, lane), acc);
#pragma unroll
                for (int r = 0; r < 8; ++r) gates[(mb + r) * 2048 + n] = acc[r];
            }
            __threadfence();
            __syncthreads();
            // elementwise: 32*512 cells
            for (int p = tid; p < 32 * 512; p += 512) {
                int b = p >> 9, j = p & 511;
                float gi = gates[b * 2048 + j];
                float gf = gates[b * 2048 + 512 + j];
                float gg = gates[b * 2048 + 1024 + j];
                float go = gates[b * 2048 + 1536 + j];
                float c2 = sigm(gf) * cdec[l * 32 * 512 + p] + sigm(gi) * tanhf(gg);
                float h2 = sigm(go) * tanhf(c2);
                cdec[l * 32 * 512 + p] = c2;
                h3[(size_t)l * 32 * 512 + p] = (half_t)h2;
                if (l == 2) hTop[(size_t)(t * BB + b) * HH + j] = (half_t)h2;
            }
            __syncthreads();
        }
    }
}

// ---------------------------------------------------------------------------
// Output head: out[b,t,n] = hTop[t*32+b] . W_out[n] + b_out[n]
// M=2016, N=32000, K=512.
// 2x5 register tiling per wave: each wave computes 2 M-tiles x 5 N-tiles,
// reusing A fragments 5x and B fragments 2x (~21 FLOP/byte from L2).
// Block = 8 waves -> 32 rows x 640 cols; grid (32000/640=50, 2016/32=63).
// ---------------------------------------------------------------------------
__global__ __launch_bounds__(256)
void head_gemm_kernel(const half_t* __restrict__ A, const half_t* __restrict__ Bt,
                      const float* __restrict__ bias, float* __restrict__ out) {
    int lane = threadIdx.x & 31;
    int wave = threadIdx.x >> 5;
    int mt0 = blockIdx.y * 32;                 // 2 M-tiles: mt0, mt0+16
    int nt0 = (blockIdx.x * 8 + wave) * 80;    // 5 N-tiles: nt0 + 16*ni
    int nl  = lane & 15;
    int mb  = (lane >> 4) * 8;

    v8f acc[2][5];
#pragma unroll
    for (int ni = 0; ni < 5; ++ni) {
        float bv = bias[nt0 + ni * 16 + nl];
#pragma unroll
        for (int mi = 0; mi < 2; ++mi)
#pragma unroll
            for (int r = 0; r < 8; ++r) acc[mi][ni][r] = bv;
    }

#pragma unroll 2
    for (int kb = 0; kb < 512; kb += 32) {
        if (kb + 32 < 512)   // compile-time foldable guard (constant trip count)
            __builtin_prefetch(Bt + (size_t)(nt0 + nl) * 512 + kb + 32, 0, 1);
        v16h a0 = load_a_tile(A, 512, mt0,      kb, lane);
        v16h a1 = load_a_tile(A, 512, mt0 + 16, kb, lane);
#pragma unroll
        for (int ni = 0; ni < 5; ++ni) {
            v16h b = load_b_tile(Bt, 512, nt0 + ni * 16, kb, lane);
            acc[0][ni] = wmma16x16x32(a0, b, acc[0][ni]);
            acc[1][ni] = wmma16x16x32(a1, b, acc[1][ni]);
        }
    }

#pragma unroll
    for (int mi = 0; mi < 2; ++mi)
#pragma unroll
        for (int ni = 0; ni < 5; ++ni)
#pragma unroll
            for (int r = 0; r < 8; ++r) {
                int m  = mt0 + mi * 16 + mb + r;   // row = t*32 + b
                int tt = m >> 5, bb_ = m & 31;
                out[(size_t)bb_ * (TDEC * VTOC) + (size_t)tt * VTOC + nt0 + ni * 16 + nl] =
                    acc[mi][ni][r];
            }
}

// ---------------------------------------------------------------------------
// Host orchestration
// ---------------------------------------------------------------------------
extern "C" void kernel_launch(void* const* d_in, const int* in_sizes, int n_in,
                              void* d_out, int out_size, void* d_ws, size_t ws_size,
                              hipStream_t stream) {
    (void)in_sizes; (void)n_in; (void)out_size; (void)ws_size;
    const int*   x        = (const int*)d_in[0];
    const int*   y        = (const int*)d_in[1];
    const float* src_emb  = (const float*)d_in[2];
    const float* tgt_emb  = (const float*)d_in[3];
    const float* W_ih_e0  = (const float*)d_in[4];
    const float* W_hh_e0  = (const float*)d_in[5];
    const float* b_e0     = (const float*)d_in[6];
    const float* W_ih_e12 = (const float*)d_in[7];
    const float* W_hh_e12 = (const float*)d_in[8];
    const float* b_e12    = (const float*)d_in[9];
    const float* W_ih_d0  = (const float*)d_in[10];
    const float* W_hh_d0  = (const float*)d_in[11];
    const float* b_d0     = (const float*)d_in[12];
    const float* W_ih_d12 = (const float*)d_in[13];
    const float* W_hh_d12 = (const float*)d_in[14];
    const float* b_d12    = (const float*)d_in[15];
    const float* W_a      = (const float*)d_in[16];
    const float* W_out    = (const float*)d_in[17];
    const float* b_out    = (const float*)d_in[18];
    float* out = (float*)d_out;

    char* ws = (char*)d_ws;
    size_t off = 0;
    auto alloc = [&](size_t bytes) -> char* {
        off = (off + 255) & ~(size_t)255;
        char* p = ws + off;
        off += bytes;
        return p;
    };
    // f16 weights
    half_t* wih_e0  = (half_t*)alloc((size_t)2 * 1024 * 256 * 2);
    half_t* whh_e0  = (half_t*)alloc((size_t)2 * 1024 * 256 * 2);
    half_t* wih_e12 = (half_t*)alloc((size_t)4 * 1024 * 512 * 2);
    half_t* whh_e12 = (half_t*)alloc((size_t)4 * 1024 * 256 * 2);
    half_t* wih_d0  = (half_t*)alloc((size_t)2048 * 1024 * 2);
    half_t* whh_d0  = (half_t*)alloc((size_t)2048 * 512 * 2);
    half_t* wih_d12 = (half_t*)alloc((size_t)2 * 2048 * 512 * 2);
    half_t* whh_d12 = (half_t*)alloc((size_t)2 * 2048 * 512 * 2);
    half_t* waT     = (half_t*)alloc((size_t)512 * 512 * 2);
    half_t* wout    = (half_t*)alloc((size_t)VTOC * 512 * 2);
    // activations
    half_t* seq0   = (half_t*)alloc((size_t)SS * BB * DD * 2);   // embedded src
    half_t* seqL1  = (half_t*)alloc((size_t)SS * BB * HH * 2);
    half_t* seqL2  = (half_t*)alloc((size_t)SS * BB * HH * 2);
    half_t* seqEnc = (half_t*)alloc((size_t)SS * BB * HH * 2);
    float*  pre_ih = (float*)alloc((size_t)2 * SS * BB * 1024 * 4);
    float*  hdec   = (float*)alloc((size_t)3 * 32 * 512 * 4);
    float*  cdec   = (float*)alloc((size_t)3 * 32 * 512 * 4);
    float*  gates  = (float*)alloc((size_t)32 * 2048 * 4);
    half_t* hTop   = (half_t*)alloc((size_t)MROW * HH * 2);

    auto cvt = [&](const float* s, half_t* d, int n) {
        cvt_f16_kernel<<<(n + 255) / 256, 256, 0, stream>>>(s, d, n);
    };
    cvt(W_ih_e0,  wih_e0,  2 * 1024 * 256);
    cvt(W_hh_e0,  whh_e0,  2 * 1024 * 256);
    cvt(W_ih_e12, wih_e12, 4 * 1024 * 512);
    cvt(W_hh_e12, whh_e12, 4 * 1024 * 256);
    cvt(W_ih_d0,  wih_d0,  2048 * 1024);
    cvt(W_hh_d0,  whh_d0,  2048 * 512);
    cvt(W_ih_d12, wih_d12, 2 * 2048 * 512);
    cvt(W_hh_d12, whh_d12, 2 * 2048 * 512);
    cvt(W_out,    wout,    VTOC * 512);
    cvt_f16_T_kernel<<<(512 * 512 + 255) / 256, 256, 0, stream>>>(W_a, waT, 512, 512);

    // source embedding gather -> f16 time-major [S,B,D]
    embed_src_kernel<<<dim3(SS, BB), DD, 0, stream>>>(x, src_emb, seq0);

    const size_t encLDS = 32 * 256 * 2 + 32 * 1024 * 4 + 32 * 256 * 4; // 180224
    // ---- encoder layer 0 ----
    for (int dir = 0; dir < 2; ++dir) {
        gemm_bt_kernel<<<dim3(1024 / 16 / 8, SS * BB / 16), 256, 0, stream>>>(
            seq0, wih_e0 + (size_t)dir * 1024 * 256, b_e0 + dir * 1024,
            pre_ih + (size_t)dir * SS * BB * 1024, SS * BB, 1024, 256);
    }
    enc_rec_kernel<<<2, 256, encLDS, stream>>>(pre_ih, whh_e0, seqL1,
                                               hdec + 0 * 32 * 512, cdec + 0 * 32 * 512);
    // ---- encoder layers 1..2 ----
    half_t* seq_in[2]   = { seqL1, seqL2 };
    half_t* seq_outb[2] = { seqL2, seqEnc };
    for (int l = 0; l < 2; ++l) {
        for (int dir = 0; dir < 2; ++dir) {
            int w = l * 2 + dir;
            gemm_bt_kernel<<<dim3(1024 / 16 / 8, SS * BB / 16), 256, 0, stream>>>(
                seq_in[l], wih_e12 + (size_t)w * 1024 * 512, b_e12 + w * 1024,
                pre_ih + (size_t)dir * SS * BB * 1024, SS * BB, 1024, 512);
        }
        enc_rec_kernel<<<2, 256, encLDS, stream>>>(
            pre_ih, whh_e12 + (size_t)l * 2 * 1024 * 256, seq_outb[l],
            hdec + (size_t)(l + 1) * 32 * 512, cdec + (size_t)(l + 1) * 32 * 512);
    }

    // ---- persistent decoder ----
    const size_t decLDS = 3 * 32 * 512 * 2 + 32 * 1024 * 2 + 32 * 512 * 2 + 32 * 128 * 4; // 212992
    dec_run_kernel<<<1, 512, decLDS, stream>>>(
        y, tgt_emb, seqEnc, waT, wih_d0, whh_d0, b_d0,
        wih_d12, whh_d12, b_d12, hdec, cdec, gates, hTop);

    // ---- output head: 2x5-tiled WMMA GEMM -> d_out [B,T-1,VT] ----
    head_gemm_kernel<<<dim3(50, 63), 256, 0, stream>>>(hTop, wout, b_out, out);
}
